// SimilarityMapBlock_4243427689203
// MI455X (gfx1250) — matrix operand order repaired and checked by reading, not compiled, not run on hardware
//
#include <hip/hip_runtime.h>
#include <hip/hip_bf16.h>
#include <stdint.h>

typedef __attribute__((ext_vector_type(16))) __bf16 v16bf;
typedef __attribute__((ext_vector_type(8)))  __bf16 v8bf;
typedef __attribute__((ext_vector_type(8)))  float  v8f;
typedef __attribute__((ext_vector_type(4)))  int    v4i;

#define N_POS 4096
#define C_IN  256
#define ICH   128
#define HGT   64
#define WID   64
#define BATCH 4

#define AS1 __attribute__((address_space(1)))
#define AS3 __attribute__((address_space(3)))

typedef AS1 v4i* gv4i_p;   // global (device) int4*
typedef AS3 v4i* lv4i_p;   // LDS int4*

#if defined(__has_builtin)
#if __has_builtin(__builtin_amdgcn_global_load_async_to_lds_b128) && \
    __has_builtin(__builtin_amdgcn_s_wait_asynccnt)
#define HAVE_ASYNC_LDS 1
#endif
#endif

// ---------------------------------------------------------------------------
// Projection: T[b][n][o] = sum_c W[o][c] * X[b][c][n] + bias[o]   (bf16 out)
// M = n positions, N = o channels, K = c. Block = 64 n-positions x all 128 o.
// 8 waves (one o-tile each) hold their W fragments (8 x v16bf) in registers
// and sweep 4 n-subtiles from one LDS-transposed stage -> 32 WMMAs per wave.
// grid (N/64, B), block 256
// ---------------------------------------------------------------------------
__global__ __launch_bounds__(256) void proj_bf16_wmma(
    const float* __restrict__ X, const float* __restrict__ W,
    const float* __restrict__ bias, __bf16* __restrict__ T)
{
    __shared__ __bf16 As[64 * 264];   // [m][c], padded row stride
    const int tid = threadIdx.x;
    const int b  = blockIdx.y;
    const int n0 = blockIdx.x * 64;

    // stage transposed: As[m][c] = X[b][c][n0+m], vectorized float4 reads
    for (int idx = tid; idx < 256 * 16; idx += 256) {
        int c = idx >> 4, v = idx & 15;
        const float4 f =
            *(const float4*)(X + ((size_t)b * C_IN + c) * N_POS + n0 + v * 4);
        As[(v * 4 + 0) * 264 + c] = (__bf16)f.x;
        As[(v * 4 + 1) * 264 + c] = (__bf16)f.y;
        As[(v * 4 + 2) * 264 + c] = (__bf16)f.z;
        As[(v * 4 + 3) * 264 + c] = (__bf16)f.w;
    }
    __syncthreads();

    const int lane = tid & 31, wave = tid >> 5;
    const int half = lane >> 4, col = lane & 15;
    const int o0 = wave * 16;

    // B fragments (weights) for all 8 K-steps, kept in registers
    v16bf wf[8];
#pragma unroll
    for (int kk = 0; kk < 8; ++kk) {
        const float* wrow = W + (size_t)(o0 + col) * C_IN + kk * 32 + half * 16;
#pragma unroll
        for (int t = 0; t < 16; ++t) wf[kk][t] = (__bf16)wrow[t];
    }
    const float bv = bias[o0 + col];

    for (int nt = 0; nt < 4; ++nt) {
        v8f acc;
#pragma unroll
        for (int r = 0; r < 8; ++r) acc[r] = bv;
#pragma unroll
        for (int kk = 0; kk < 8; ++kk) {
            const int c0 = kk * 32;
            const __bf16* arow = &As[(nt * 16 + col) * 264];
            const v8bf lo = *(const v8bf*)(arow + c0 + half * 16);
            const v8bf hi = *(const v8bf*)(arow + c0 + half * 16 + 8);
            v16bf a;
#pragma unroll
            for (int t = 0; t < 8; ++t) { a[t] = lo[t]; a[8 + t] = hi[t]; }
            acc = __builtin_amdgcn_wmma_f32_16x16x32_bf16(
                false, a, false, wf[kk], (short)0, acc, false, false);
        }
#pragma unroll
        for (int r = 0; r < 8; ++r) {
            int m = nt * 16 + half * 8 + r;
            T[((size_t)b * N_POS + n0 + m) * ICH + o0 + col] = (__bf16)acc[r];
        }
    }
}

// ---------------------------------------------------------------------------
// Stage one 64-row phi chunk (64 x 128 bf16) into LDS (row stride 144).
// Async GLOBAL_LOAD_ASYNC_TO_LDS_B128 when available, else plain copies.
// ---------------------------------------------------------------------------
__device__ __forceinline__ void stage_chunk(const __bf16* __restrict__ g,
                                            __bf16* l, int tid)
{
    for (int idx = tid; idx < 64 * 16; idx += 256) {
        const int row = idx >> 4, seg = idx & 15;
        const __bf16* src = g + row * 128 + seg * 8;
        __bf16* dst = l + row * 144 + seg * 8;
#if HAVE_ASYNC_LDS
        __builtin_amdgcn_global_load_async_to_lds_b128(
            (gv4i_p)(uintptr_t)src,
            (lv4i_p)(uint32_t)(uintptr_t)dst, 0, 0);
#else
        *(v8bf*)dst = *(const v8bf*)src;
#endif
    }
}

__device__ __forceinline__ void wait_async_done()
{
#if HAVE_ASYNC_LDS
    __builtin_amdgcn_s_wait_asynccnt(0);
#endif
}

// ---------------------------------------------------------------------------
// Correlation + fused row-max: sim[b][i] = max_j sum_c theta[i][c]*phi[j][c]
// Block = 128 query rows (8 waves, one 16-row i-tile each). The phi stream is
// shared: 64-row chunks double-buffered through LDS, so phi L2 traffic is
// 1 MB per block instead of 1 MB per i-tile (8x less). 16 WMMA per chunk
// per wave; row-max folded into the accumulator tiles, never materializing f.
// grid (N/128, B), block 256
// ---------------------------------------------------------------------------
__global__ __launch_bounds__(256) void corr_rowmax_wmma(
    const __bf16* __restrict__ Tq, const __bf16* __restrict__ Tk,
    float* __restrict__ sim)
{
    __shared__ __bf16 Bs[2][64 * 144];   // double-buffered chunk, padded rows
    const int tid = threadIdx.x;
    const int b  = blockIdx.y;
    const int lane = tid & 31, wave = tid >> 5;
    const int half = lane >> 4, col = lane & 15;
    const int i0 = blockIdx.x * 128 + wave * 16;

    // A fragments: theta rows i0..i0+15, all K = 0..127, held in registers
    const __bf16* arow = Tq + ((size_t)b * N_POS + i0 + col) * ICH;
    v16bf a[4];
#pragma unroll
    for (int kk = 0; kk < 4; ++kk) {
        const v8bf lo = *(const v8bf*)(arow + kk * 32 + half * 16);
        const v8bf hi = *(const v8bf*)(arow + kk * 32 + half * 16 + 8);
#pragma unroll
        for (int t = 0; t < 8; ++t) { a[kk][t] = lo[t]; a[kk][8 + t] = hi[t]; }
    }

    v8f rmax;
#pragma unroll
    for (int r = 0; r < 8; ++r) rmax[r] = -3.4e38f;

    const __bf16* phi_b = Tk + (size_t)b * N_POS * ICH;
    stage_chunk(phi_b, &Bs[0][0], tid);

    for (int cj = 0; cj < 64; ++cj) {
        const int cur = cj & 1;
        wait_async_done();       // chunk cj resident in Bs[cur]
        __syncthreads();         // all waves past previous compute + see data
        if (cj + 1 < 64)
            stage_chunk(phi_b + (size_t)(cj + 1) * 64 * ICH, &Bs[1 - cur][0], tid);

        const __bf16* bs = &Bs[cur][0];
#pragma unroll
        for (int t = 0; t < 4; ++t) {
            v8f acc = {0.f, 0.f, 0.f, 0.f, 0.f, 0.f, 0.f, 0.f};
#pragma unroll
            for (int kk = 0; kk < 4; ++kk) {
                const __bf16* p = bs + (t * 16 + col) * 144 + kk * 32 + half * 16;
                v16bf bm = *(const v16bf*)p;   // 32B ds load, 32B aligned
                acc = __builtin_amdgcn_wmma_f32_16x16x32_bf16(
                    false, a[kk], false, bm, (short)0, acc, false, false);
            }
#pragma unroll
            for (int r = 0; r < 8; ++r) rmax[r] = fmaxf(rmax[r], acc[r]);
        }
    }

    // reduce over the 16 columns within each 16-lane half; wave owns its rows
#pragma unroll
    for (int r = 0; r < 8; ++r) {
        float v = rmax[r];
        v = fmaxf(v, __shfl_xor(v, 1, 32));
        v = fmaxf(v, __shfl_xor(v, 2, 32));
        v = fmaxf(v, __shfl_xor(v, 4, 32));
        v = fmaxf(v, __shfl_xor(v, 8, 32));
        rmax[r] = v;
    }
    if (col == 0) {
#pragma unroll
        for (int r = 0; r < 8; ++r)
            sim[(size_t)b * N_POS + i0 + half * 8 + r] = rmax[r];
    }
}

// ---------------------------------------------------------------------------
// 3x3 conv with reflection pad 1 (NCHW / OIHW), optional residual add + relu.
// grid (W/16, H/16, B), block (16,16). Dynamic LDS: input halo tile + weights.
// ---------------------------------------------------------------------------
__device__ __forceinline__ int refl(int p) {
    return p < 0 ? -p : (p > 63 ? 126 - p : p);
}

__global__ __launch_bounds__(256) void conv3x3_refl(
    const float* __restrict__ in, int ICn,
    const float* __restrict__ w, const float* __restrict__ bias, int OCn,
    const float* __restrict__ addsrc, int relu, float* __restrict__ out)
{
    extern __shared__ float smem[];
    float* sIn = smem;               // ICn * 18*18
    float* sW  = smem + ICn * 324;   // OCn * ICn * 9

    const int tx = threadIdx.x, ty = threadIdx.y;
    const int tid = ty * 16 + tx;
    const int x0 = blockIdx.x * 16, y0 = blockIdx.y * 16, b = blockIdx.z;

    for (int idx = tid; idx < ICn * 324; idx += 256) {
        int ic = idx / 324, p = idx % 324;
        int py = p / 18, px = p % 18;
        int gy = refl(y0 - 1 + py), gx = refl(x0 - 1 + px);
        sIn[idx] = in[(((size_t)b * ICn + ic) * HGT + gy) * WID + gx];
    }
    for (int idx = tid; idx < OCn * ICn * 9; idx += 256) sW[idx] = w[idx];
    __syncthreads();

    float acc[32];
    for (int oc = 0; oc < OCn; ++oc) acc[oc] = 0.f;

    for (int ic = 0; ic < ICn; ++ic) {
        float v[9];
        const float* base = sIn + ic * 324 + ty * 18 + tx;
#pragma unroll
        for (int t = 0; t < 9; ++t) v[t] = base[(t / 3) * 18 + (t % 3)];
        const float* wr = sW + ic * 9;
        for (int oc = 0; oc < OCn; ++oc) {
            const float* wk = wr + (size_t)oc * ICn * 9;
            float s = 0.f;
#pragma unroll
            for (int t = 0; t < 9; ++t) s = fmaf(wk[t], v[t], s);
            acc[oc] += s;
        }
    }

    const int y = y0 + ty, x = x0 + tx;
    for (int oc = 0; oc < OCn; ++oc) {
        float vv = acc[oc] + bias[oc];
        size_t oidx = (((size_t)b * OCn + oc) * HGT + y) * WID + x;
        if (addsrc) vv += addsrc[oidx];
        if (relu) vv = fmaxf(vv, 0.f);
        out[oidx] = vv;
    }
}

// ---------------------------------------------------------------------------
extern "C" void kernel_launch(void* const* d_in, const int* in_sizes, int n_in,
                              void* d_out, int out_size, void* d_ws, size_t ws_size,
                              hipStream_t stream)
{
    (void)in_sizes; (void)n_in; (void)out_size; (void)ws_size;
    const float* q       = (const float*)d_in[0];
    const float* kf      = (const float*)d_in[1];
    const float* w_theta = (const float*)d_in[2];
    const float* b_theta = (const float*)d_in[3];
    const float* w_phi   = (const float*)d_in[4];
    const float* b_phi   = (const float*)d_in[5];
    const float* w_in    = (const float*)d_in[6];
    const float* b_in    = (const float*)d_in[7];
    const float* w_r1    = (const float*)d_in[8];
    const float* b_r1    = (const float*)d_in[9];
    const float* w_r2    = (const float*)d_in[10];
    const float* b_r2    = (const float*)d_in[11];
    const float* w_out   = (const float*)d_in[12];
    const float* b_out   = (const float*)d_in[13];

    char* ws = (char*)d_ws;
    __bf16* theta = (__bf16*)(ws);                 // 4 MiB
    __bf16* phi   = (__bf16*)(ws + (4u  << 20));   // 4 MiB
    float*  sim   = (float*)(ws + (8u  << 20));    // 64 KiB
    float*  h     = (float*)(ws + (9u  << 20));    // 2 MiB
    float*  r     = (float*)(ws + (11u << 20));    // 2 MiB
    float*  hs    = (float*)(ws + (13u << 20));    // 2 MiB
    float*  out   = (float*)d_out;

    dim3 pblk(256), pgrd(N_POS / 64, BATCH);
    hipLaunchKernelGGL(proj_bf16_wmma, pgrd, pblk, 0, stream, q,  w_theta, b_theta, theta);
    hipLaunchKernelGGL(proj_bf16_wmma, pgrd, pblk, 0, stream, kf, w_phi,   b_phi,   phi);

    hipLaunchKernelGGL(corr_rowmax_wmma, dim3(N_POS / 128, BATCH), dim3(256), 0, stream,
                       theta, phi, sim);

    dim3 cblk(16, 16), cgrd(WID / 16, HGT / 16, BATCH);
    size_t sm1 = (size_t)(1  * 324 + 32 * 1  * 9) * sizeof(float);
    size_t sm2 = (size_t)(32 * 324 + 32 * 32 * 9) * sizeof(float);
    size_t sm4 = (size_t)(32 * 324 + 1  * 32 * 9) * sizeof(float);
    hipLaunchKernelGGL(conv3x3_refl, cgrd, cblk, sm1, stream, sim, 1,  w_in,  b_in,  32, (const float*)nullptr, 0, h);
    hipLaunchKernelGGL(conv3x3_refl, cgrd, cblk, sm2, stream, h,   32, w_r1,  b_r1,  32, (const float*)nullptr, 1, r);
    hipLaunchKernelGGL(conv3x3_refl, cgrd, cblk, sm2, stream, r,   32, w_r2,  b_r2,  32, h,                     0, hs);
    hipLaunchKernelGGL(conv3x3_refl, cgrd, cblk, sm4, stream, hs,  32, w_out, b_out, 1,  (const float*)nullptr, 0, out);
}
